// MPNN_79645873537465
// MI455X (gfx1250) — matrix operand order repaired and checked by reading, not compiled, not run on hardware
//
#include <hip/hip_runtime.h>

typedef __bf16 bf16_t;
typedef __attribute__((ext_vector_type(16))) __bf16 v16bf;
typedef __attribute__((ext_vector_type(8)))  __bf16 v8bf;
typedef __attribute__((ext_vector_type(2)))  __bf16 v2bf;
typedef __attribute__((ext_vector_type(8)))  float  v8f;

#define DEMB  64
#define WCOLS 192   // [M1 | M0 | root]
#define NTILES (WCOLS / 16)
#define NLAYERS 3

// ---------------- elementwise helpers ----------------

__global__ void zero_f32_kernel(float* __restrict__ p, int n) {
    int t = blockIdx.x * blockDim.x + threadIdx.x;
    if (t < n) p[t] = 0.0f;
}

// Xr[n,o] = bf16(relu(embed(bipartite inputs)));  one thread = 2 adjacent outputs
__global__ void embed_kernel(const float* __restrict__ xu, const float* __restrict__ xv,
                             const float* __restrict__ Wu, const float* __restrict__ bu,
                             const float* __restrict__ Wv, const float* __restrict__ bv,
                             bf16_t* __restrict__ Xr, int Nu, int N) {
    int tid = blockIdx.x * blockDim.x + threadIdx.x;   // over N*32
    if (tid >= N * (DEMB / 2)) return;
    int n = tid >> 5;
    int o = (tid & 31) * 2;
    float v0, v1;
    if (n < Nu) {
        float a = xu[n];
        v0 = a * Wu[o]     + bu[o];
        v1 = a * Wu[o + 1] + bu[o + 1];
    } else {
        int m = n - Nu;
        float a0 = xv[2 * m], a1 = xv[2 * m + 1];
        v0 = a0 * Wv[o]     + a1 * Wv[DEMB + o]     + bv[o];
        v1 = a0 * Wv[o + 1] + a1 * Wv[DEMB + o + 1] + bv[o + 1];
    }
    v2bf r = { (bf16_t)fmaxf(v0, 0.0f), (bf16_t)fmaxf(v1, 0.0f) };
    *(v2bf*)(Xr + (size_t)n * DEMB + o) = r;
}

// Wt[o][k] = Wcat[k][o] in bf16; Wcat cols 0-63 = M1, 64-127 = M0, 128-191 = root
__global__ void wprep_kernel(const float* __restrict__ Wl1, const float* __restrict__ bl1,
                             const float* __restrict__ root, bf16_t* __restrict__ Wt) {
    int tid = blockIdx.x * blockDim.x + threadIdx.x;
    if (tid >= WCOLS * DEMB) return;
    int o = tid >> 6, k = tid & 63;
    float v;
    if (o < 64)       v = Wl1[k * 64 + o];
    else if (o < 128) v = bl1[k * 64 + (o - 64)];
    else              v = root[k * 64 + (o - 128)];
    Wt[tid] = (bf16_t)v;
}

__global__ void deg_kernel(const int* __restrict__ ei, int* __restrict__ deg, int E) {
    int e = blockIdx.x * blockDim.x + threadIdx.x;
    if (e >= E) return;
    atomicAdd(&deg[ei[E + e]], 1);
}

// in-place: int degree -> float 1/max(deg,1)
__global__ void invdenom_kernel(int* __restrict__ deg, int N) {
    int n = blockIdx.x * blockDim.x + threadIdx.x;
    if (n >= N) return;
    int d = deg[n];
    ((float*)deg)[n] = 1.0f / fmaxf((float)d, 1.0f);
}

// ---------------- WMMA GEMM: Y[N,192] = Xr[N,64] @ Wt^T ----------------
// One wave owns a 16x192 row-strip: A fragments loaded once, 12 B tiles,
// 24 x v_wmma_f32_16x16x32_bf16 per wave.
__global__ void __launch_bounds__(256) gemm_kernel(const bf16_t* __restrict__ Xr,
                                                   const bf16_t* __restrict__ Wt,
                                                   float* __restrict__ Y, int tilesM) {
    int wave = blockIdx.x * (blockDim.x >> 5) + (threadIdx.x >> 5);
    if (wave >= tilesM) return;               // wave-uniform: EXEC stays all-1s
    int lane = threadIdx.x & 31;
    int l  = lane & 15;
    int hi = lane >> 4;

    // ---- load both K-block A fragments once (row M = l of the strip) ----
    const bf16_t* Arow = Xr + (size_t)(wave * 16 + l) * DEMB;
    v16bf av[2];
#pragma unroll
    for (int kk = 0; kk < 2; ++kk) {
        int kb = kk * 32;
        // elems 0..15 = K kb+hi*16 .. +15 (contiguous, matches B)
        v8bf a0 = *(const v8bf*)(Arow + kb + hi * 16);
        v8bf a1 = *(const v8bf*)(Arow + kb + hi * 16 + 8);
#pragma unroll
        for (int j = 0; j < 8; ++j) { av[kk][j] = a0[j]; av[kk][8 + j] = a1[j]; }
    }

    float* Ybase = Y + (size_t)(wave * 16 + hi * 8) * WCOLS + l;

    // ---- sweep the 12 output tiles; B (24 KB total) stays cache-resident ----
#pragma unroll
    for (int tN = 0; tN < NTILES; ++tN) {
        const bf16_t* Brow = Wt + (size_t)(tN * 16 + l) * DEMB;   // column N = l
        v8f acc = {};
#pragma unroll
        for (int kk = 0; kk < 2; ++kk) {
            // B frag: elems j = K kk*32 + hi*16 + j (contiguous in transposed Wt)
            v16bf bv = *(const v16bf*)(Brow + kk * 32 + hi * 16);
            acc = __builtin_amdgcn_wmma_f32_16x16x32_bf16(false, av[kk], false, bv,
                                                          (short)0, acc, false, false);
        }
        // C/D layout: VGPR r -> row hi*8 + r, col l
        float* Yout = Ybase + tN * 16;
#pragma unroll
        for (int r = 0; r < 8; ++r) Yout[(size_t)r * WCOLS] = acc[r];
    }
}

// ---------------- edge scatter: accum[dst] += a_e*y1[src] + y0[src] ----------------
__global__ void __launch_bounds__(256) scatter_kernel(const int* __restrict__ ei,
                                                      const float* __restrict__ ea,
                                                      const float* __restrict__ Y,
                                                      float* __restrict__ accum, int E) {
    int wave = blockIdx.x * (blockDim.x >> 5) + (threadIdx.x >> 5);
    if (wave >= E) return;
    int lane = threadIdx.x & 31;
    int src = ei[wave];
    int dst = ei[E + wave];
    float a = ea[wave];
    const float* y = Y + (size_t)src * WCOLS;
    float* out = accum + (size_t)dst * DEMB;
    float m0 = a * y[lane]      + y[64 + lane];   // a*y1 + y0, outputs [0,32)
    float m1 = a * y[32 + lane] + y[96 + lane];   // outputs [32,64)
    unsafeAtomicAdd(out + lane, m0);              // global_atomic_add_f32
    unsafeAtomicAdd(out + 32 + lane, m1);
}

// intermediate layers: Xr = bf16(relu(accum*inv_denom + Yroot + b_conv))
__global__ void finalize_mid_kernel(const float* __restrict__ accum,
                                    const float* __restrict__ Y,
                                    const float* __restrict__ inv_denom,
                                    const float* __restrict__ bconv,
                                    bf16_t* __restrict__ Xr, int N) {
    int tid = blockIdx.x * blockDim.x + threadIdx.x;   // over N*32
    if (tid >= N * (DEMB / 2)) return;
    int n = tid >> 5;
    int o = (tid & 31) * 2;
    float inv = inv_denom[n];
    const float* yr = Y + (size_t)n * WCOLS + 128;
    const float* ac = accum + (size_t)n * DEMB;
    float v0 = ac[o]     * inv + yr[o]     + bconv[o];
    float v1 = ac[o + 1] * inv + yr[o + 1] + bconv[o + 1];
    v2bf r = { (bf16_t)fmaxf(v0, 0.0f), (bf16_t)fmaxf(v1, 0.0f) };
    *(v2bf*)(Xr + (size_t)n * DEMB + o) = r;
}

// last layer: f32 output, no relu
__global__ void finalize_last_kernel(const float* __restrict__ accum,
                                     const float* __restrict__ Y,
                                     const float* __restrict__ inv_denom,
                                     const float* __restrict__ bconv,
                                     float* __restrict__ out, int N) {
    int tid = blockIdx.x * blockDim.x + threadIdx.x;   // over N*32
    if (tid >= N * (DEMB / 2)) return;
    int n = tid >> 5;
    int o = (tid & 31) * 2;
    float inv = inv_denom[n];
    const float* yr = Y + (size_t)n * WCOLS + 128;
    const float* ac = accum + (size_t)n * DEMB;
    float2 r;
    r.x = ac[o]     * inv + yr[o]     + bconv[o];
    r.y = ac[o + 1] * inv + yr[o + 1] + bconv[o + 1];
    *(float2*)(out + (size_t)n * DEMB + o) = r;
}

// ---------------- host ----------------

extern "C" void kernel_launch(void* const* d_in, const int* in_sizes, int n_in,
                              void* d_out, int out_size, void* d_ws, size_t ws_size,
                              hipStream_t stream) {
    const float* xu    = (const float*)d_in[0];
    const float* xv    = (const float*)d_in[1];
    const int*   ei    = (const int*)d_in[2];
    const float* ea    = (const float*)d_in[3];
    // d_in[4] = i (==3 -> 3 layers), d_in[5] = dummy : unused on device
    const float* Wl1   = (const float*)d_in[6];
    const float* bl1   = (const float*)d_in[7];
    const float* Wu    = (const float*)d_in[8];
    const float* bu    = (const float*)d_in[9];
    const float* Wv    = (const float*)d_in[10];
    const float* bv    = (const float*)d_in[11];
    const float* root  = (const float*)d_in[12];
    const float* bconv = (const float*)d_in[13];

    const int Nu = in_sizes[0];
    const int Nv = in_sizes[1] / 2;
    const int N  = Nu + Nv;
    const int E  = in_sizes[3];

    // workspace carve-out
    char* ws = (char*)d_ws;
    size_t off = 0;
    auto carve = [&](size_t bytes) -> void* {
        void* p = ws + off;
        off += (bytes + 255) & ~(size_t)255;
        return p;
    };
    bf16_t* Xr    = (bf16_t*)carve((size_t)N * DEMB * sizeof(bf16_t));
    float*  Y     = (float*)carve((size_t)N * WCOLS * sizeof(float));
    float*  accum = (float*)carve((size_t)N * DEMB * sizeof(float));
    int*    deg   = (int*)carve((size_t)N * sizeof(int));   // becomes inv_denom (f32)
    bf16_t* Wt    = (bf16_t*)carve((size_t)WCOLS * DEMB * sizeof(bf16_t));

    const int TB = 256;
    const int ND  = N * DEMB;
    const int ND2 = N * (DEMB / 2);
    dim3 blk(TB);

    // embeddings (fused relu+bf16 cvt), fused weights, degrees
    embed_kernel<<<(ND2 + TB - 1) / TB, blk, 0, stream>>>(xu, xv, Wu, bu, Wv, bv, Xr, Nu, N);
    wprep_kernel<<<(WCOLS * DEMB + TB - 1) / TB, blk, 0, stream>>>(Wl1, bl1, root, Wt);
    zero_f32_kernel<<<(N + TB - 1) / TB, blk, 0, stream>>>((float*)deg, N);  // int 0 == f32 0 bits
    deg_kernel<<<(E + TB - 1) / TB, blk, 0, stream>>>(ei, deg, E);
    invdenom_kernel<<<(N + TB - 1) / TB, blk, 0, stream>>>(deg, N);
    const float* inv_denom = (const float*)deg;

    const int tilesM   = N / 16;                 // one wave per 16-row strip
    const int gemmGrid = (tilesM + 7) / 8;       // 8 waves per 256-thread block
    const int scatGrid = (E + 7) / 8;

    for (int layer = 0; layer < NLAYERS; ++layer) {
        zero_f32_kernel<<<(ND + TB - 1) / TB, blk, 0, stream>>>(accum, ND);
        gemm_kernel<<<gemmGrid, blk, 0, stream>>>(Xr, Wt, Y, tilesM);
        scatter_kernel<<<scatGrid, blk, 0, stream>>>(ei, ea, Y, accum, E);
        if (layer == NLAYERS - 1) {
            finalize_last_kernel<<<(ND2 + TB - 1) / TB, blk, 0, stream>>>(
                accum, Y, inv_denom, bconv, (float*)d_out, N);
        } else {
            finalize_mid_kernel<<<(ND2 + TB - 1) / TB, blk, 0, stream>>>(
                accum, Y, inv_denom, bconv, Xr, N);
        }
    }
}
